// SphericalFilter_50036368999144
// MI455X (gfx1250) — matrix operand order repaired and checked by reading, not compiled, not run on hardware
//
#include <hip/hip_runtime.h>
#include <math.h>

typedef __attribute__((ext_vector_type(16))) _Float16 v16h;
typedef __attribute__((ext_vector_type(8)))  float    v8f;

#define KDIM   129         // 2*64+1 harmonics
#define KP     160         // K padded to 5 * 32
#define KSTEPS 5
#define MT     128         // pixels per block
#define NT     128         // channels per block
#define CP     132         // C-tile LDS pitch (floats): 16B aligned, conflict-free
#define MAXF   64

__global__ __launch_bounds__(256)
void sph_wmma_kernel(const float* __restrict__ x,
                     const float* __restrict__ W,
                     const float* __restrict__ bvec,
                     float* __restrict__ out,
                     int npix, int nch)
{
    extern __shared__ char smem[];
    float*    cs1 = (float*)smem;                 // [KP] sqrt((2m+1)/(2m))
    float*    ca  = cs1 + KP;                     // [KP] sqrt((4(m+1)^2-1)/(2m+1))
    _Float16* Ysh = (_Float16*)(ca + KP);         // [MT][KP]  A tile, K fastest
    _Float16* Bsh = Ysh + MT * KP;                // [NT][KP]  W^T tile, K fastest
    float*    Csh = (float*)smem;                 // [MT][CP]  C tile (reuses space)

    const int tid  = threadIdx.x;
    const int lane = tid & 31;
    const int wid  = tid >> 5;

    const int nChBlocks = nch / NT;
    const int pixBase = (blockIdx.x / nChBlocks) * MT;
    const int chBase  = (blockIdx.x % nChBlocks) * NT;

    // ---- phase 0: per-m recurrence coefficients -----------------------
    if (tid <= MAXF) {
        int m = tid;
        cs1[m] = (m > 0) ? sqrtf((2.0f*m + 1.0f) / (2.0f*m)) : 0.0f;
        ca[m]  = sqrtf((4.0f*(m+1)*(m+1) - 1.0f) / (2.0f*m + 1.0f));
    }
    __syncthreads();

    // ---- phase 1: waves 0-3 build Y (f16); waves 4-7 stage W^T (f16) ----
    if (tid < MT) {
        const int p  = tid;
        const int gp = pixBase + p;
        float theta = 0.0f, phi = 0.0f;
        if (gp < npix) { theta = x[2*gp + 0]; phi = x[2*gp + 1]; }
        const float ct = __cosf(phi);
        const float st = sqrtf(fmaxf(1.0f - ct*ct, 0.0f));
        const float c1 = __cosf(theta);
        const float s1 = __sinf(theta);

        _Float16* yrow = Ysh + p * KP;
        float pmm = 0.28209479177387814f;          // sqrt(1/(4*pi))
        yrow[MAXF] = (_Float16)(ca[0] * ct * pmm); // m = 0 term (l = 1)
        float cm = 1.0f, sm = 0.0f;                // cos(m*th), sin(m*th)
        #pragma unroll 4
        for (int m = 1; m <= MAXF; ++m) {
            pmm = -cs1[m] * st * pmm;              // Condon-Shortley phase
            const float cn = cm*c1 - sm*s1;        // Chebyshev rotation
            const float sn = sm*c1 + cm*s1;
            cm = cn; sm = sn;
            const float yb  = ca[m] * ct * pmm;    // Ybar_{m+1,m}
            const float sg  = (m & 1) ? -1.41421356237309515f
                                      :  1.41421356237309515f;
            yrow[MAXF + m] = (_Float16)(sg * yb * cm);   // +m : cos branch
            yrow[MAXF - m] = (_Float16)(sg * yb * sm);   // -m : sin branch
        }
        #pragma unroll
        for (int k = KDIM; k < KP; ++k) yrow[k] = (_Float16)0.0f;  // pad K
    } else {
        for (int idx = tid - MT; idx < NT * KP; idx += 128) {
            const int c = idx / KP;
            const int k = idx - c * KP;
            float v = 0.0f;
            if (k < KDIM) v = W[(size_t)(chBase + c) * KDIM + k];
            Bsh[c * KP + k] = (_Float16)v;
        }
    }
    __syncthreads();

    // ---- phase 2: WMMA f16 -> f32; batch B-frag loads to overlap LDS ----
    const int mrow = lane & 15;            // A row / B column inside a tile
    const int hi   = (lane >> 4) & 1;      // lanes 16..31 take the high K half
    v8f acc[8] = {};

    #pragma unroll
    for (int ks = 0; ks < KSTEPS; ++ks) {
        const int kb = ks * 32;
        v16h afrag;
        {   // 16-bit A layout: lane<16 -> K 0-7 / 16-23 ; lane>=16 -> K 8-15 / 24-31
            const _Float16* src = Ysh + (wid*16 + mrow) * KP + kb + (hi ? 8 : 0);
            *(float4*)&afrag         = *(const float4*)(src);
            *(((float4*)&afrag) + 1) = *(const float4*)(src + 16);
        }
        #pragma unroll
        for (int g = 0; g < 2; ++g) {
            v16h bf[4];
            #pragma unroll
            for (int j = 0; j < 4; ++j) {
                // 16-bit B layout: lane<16 -> K 0-7 / 16-23 ; lane>=16 -> K 8-15 / 24-31
                const _Float16* src =
                    Bsh + ((g*4 + j)*16 + mrow) * KP + kb + (hi ? 8 : 0);
                *(float4*)&bf[j]         = *(const float4*)(src);
                *(((float4*)&bf[j]) + 1) = *(const float4*)(src + 16);
            }
            #pragma unroll
            for (int j = 0; j < 4; ++j)
                acc[g*4 + j] = __builtin_amdgcn_wmma_f32_16x16x32_f16(
                    false, afrag, false, bf[j], (short)0, acc[g*4 + j],
                    false, false);
        }
    }

    // ---- phase 3: bias into C tile in LDS (transpose for wide stores) ----
    float bb[8];
    #pragma unroll
    for (int ns = 0; ns < 8; ++ns) bb[ns] = bvec[chBase + ns*16 + mrow];

    __syncthreads();   // everyone done reading Ysh/Bsh before Csh overwrite

    const int mtop = hi ? 8 : 0;
    #pragma unroll
    for (int ns = 0; ns < 8; ++ns) {
        #pragma unroll
        for (int r = 0; r < 8; ++r)
            Csh[(wid*16 + mtop + r) * CP + ns*16 + mrow] = acc[ns][r] + bb[ns];
    }
    __syncthreads();

    // ---- phase 4: coalesced 128-bit global stores (512B per wave/inst) ----
    const int totalV4 = MT * (NT / 4);     // 4096 float4 per block tile
    for (int idx = tid; idx < totalV4; idx += 256) {
        const int row = idx >> 5;          // / (NT/4)
        const int c4  = idx & 31;
        const int pix = pixBase + row;
        if (pix < npix) {
            const float4 v = *(const float4*)&Csh[row * CP + c4 * 4];
            *(float4*)&out[(size_t)pix * nch + chBase + c4 * 4] = v;
        }
    }
}

extern "C" void kernel_launch(void* const* d_in, const int* in_sizes, int n_in,
                              void* d_out, int out_size, void* d_ws, size_t ws_size,
                              hipStream_t stream) {
    (void)n_in; (void)out_size; (void)d_ws; (void)ws_size;
    const float* x = (const float*)d_in[0];   // (4,128,256,1,2) f32
    const float* W = (const float*)d_in[1];   // (1024,129) f32
    const float* b = (const float*)d_in[2];   // (1024,) f32
    float* out = (float*)d_out;               // (4,128,256,1,1024) f32

    const int npix = in_sizes[0] / 2;         // 131072
    const int nch  = in_sizes[2];             // 1024

    const int nPixBlocks = (npix + MT - 1) / MT;
    const int nChBlocks  = nch / NT;
    size_t shmemAB = (size_t)(2 * KP) * sizeof(float)
                   + (size_t)(MT + NT) * KP * sizeof(_Float16);
    size_t shmemC  = (size_t)MT * CP * sizeof(float);
    size_t shmem   = shmemAB > shmemC ? shmemAB : shmemC;   // ~81 KB (<320 KB WGP LDS)

    hipLaunchKernelGGL(sph_wmma_kernel,
                       dim3(nPixBlocks * nChBlocks), dim3(256), shmem, stream,
                       x, W, b, out, npix, nch);
}
